// SwinTransformerBlock3D_43490838840018
// MI455X (gfx1250) — matrix operand order repaired and checked by reading, not compiled, not run on hardware
//
#include <hip/hip_runtime.h>
#include <hip/hip_bf16.h>

typedef _Float16 h8   __attribute__((ext_vector_type(8)));
typedef _Float16 v16h __attribute__((ext_vector_type(16)));
typedef float    v8f  __attribute__((ext_vector_type(8)));

#define TOK 65536
#define CCH 192

// windowed(+shifted) token index -> original flat spatial token index
__device__ __forceinline__ int win_to_orig(int t) {
    int b   = t >> 15;            // 32768 tokens per batch
    int r   = t & 32767;
    int win = r >> 6;             // 0..511  (= (wh*8+ww)*8+wd)
    int n   = r & 63;             // 0..63   (= (th*4+tw)*4+td)
    int wh = win >> 6, ww = (win >> 3) & 7, wd = win & 7;
    int th = n >> 4,   tw = (n >> 2) & 3,  td = n & 3;
    int oh = (wh * 4 + th + 2) & 31;   // roll(-2) forward / roll(+2) reverse
    int ow = (ww * 4 + tw + 2) & 31;
    int od = (wd * 4 + td + 2) & 31;
    return ((b * 32 + oh) * 32 + ow) * 32 + od;
}

// Build a 16-element A/B fragment from two contiguous 8-half LDS runs.
__device__ __forceinline__ v16h frag16(const _Float16* p0, const _Float16* p1) {
    h8 lo = *(const h8*)p0;
    h8 hi = *(const h8*)p1;
    v16h r;
#pragma unroll
    for (int j = 0; j < 8; ++j) { r[j] = lo[j]; r[j + 8] = hi[j]; }
    return r;
}

// ---------------------------------------------------------------------------
// One-shot weight conversion: W f32 [K][N] row-major -> Wt f16 [N][K]
// (transposed so GEMM B-fragments become two contiguous 8-half LDS runs).
// ---------------------------------------------------------------------------
__global__ __launch_bounds__(256) void wconv_kernel(
    const float* __restrict__ W, _Float16* __restrict__ Wt, int K, int N)
{
    int idx = blockIdx.x * 256 + threadIdx.x;
    if (idx >= K * N) return;
    int n = idx / K, k = idx - n * K;
    Wt[idx] = (_Float16)W[(size_t)k * N + n];
}

// ---------------------------------------------------------------------------
// LayerNorm (one wave per token). remap!=0: output token order is the
// shifted/window-partitioned order, source read from original layout.
// ---------------------------------------------------------------------------
__global__ __launch_bounds__(32) void ln_kernel(
    const float* __restrict__ x, const float* __restrict__ g,
    const float* __restrict__ bta, _Float16* __restrict__ out, int remap)
{
    int t    = blockIdx.x;
    int src  = remap ? win_to_orig(t) : t;
    int lane = threadIdx.x;
    const float* xr = x + (size_t)src * CCH;

    float vals[6];
    float s = 0.f;
#pragma unroll
    for (int j = 0; j < 6; ++j) { vals[j] = xr[lane + 32 * j]; s += vals[j]; }
#pragma unroll
    for (int o = 16; o > 0; o >>= 1) s += __shfl_xor(s, o, 32);
    float mean = s * (1.0f / 192.0f);

    float vs = 0.f;
#pragma unroll
    for (int j = 0; j < 6; ++j) { float d = vals[j] - mean; vs += d * d; }
#pragma unroll
    for (int o = 16; o > 0; o >>= 1) vs += __shfl_xor(vs, o, 32);
    float inv = rsqrtf(vs * (1.0f / 192.0f) + 1e-5f);

#pragma unroll
    for (int j = 0; j < 6; ++j) {
        int c = lane + 32 * j;
        out[(size_t)t * CCH + c] = (_Float16)((vals[j] - mean) * inv * g[c] + bta[c]);
    }
}

// ---------------------------------------------------------------------------
// 64x64-tile WMMA GEMM:  out[M,N] = A[M,K](f16) * Wt[N,K](f16, pre-transposed)
// K-step 64: each wave does 4 WMMAs per staging round (2 k-chunks x 2 ni).
// EPI 0: f16 store           (QKV projection)
// EPI 1: exact GELU, f16     (MLP fc1)
// EPI 2: window-reverse remap + residual(x), f32   (attn out projection)
// EPI 3: residual(xo) natural order, f32           (MLP fc2 -> d_out)
// ---------------------------------------------------------------------------
template <int EPI>
__global__ __launch_bounds__(256) void gemm64x64(
    const _Float16* __restrict__ A, const _Float16* __restrict__ Wg,
    const float* __restrict__ bias, _Float16* __restrict__ outh,
    float* __restrict__ outf, const float* __restrict__ resid,
    int K, int N)
{
    __shared__ __align__(16) _Float16 As[64 * 72];   // A tile  [m][k0..k0+63]
    __shared__ __align__(16) _Float16 Ws[64 * 72];   // W tile  [n][k0..k0+63]

    const int tid  = threadIdx.x;
    const int wave = tid >> 5;
    const int lane = tid & 31;
    const int half = lane >> 4;
    const int l16  = lane & 15;
    const int m0   = blockIdx.x * 64;
    const int n0   = blockIdx.y * 64;
    const int mi   = wave >> 1;          // 0..3
    const int nia  = (wave & 1) * 2;     // 0 or 2; wave owns ni = nia, nia+1

    const int sr = tid >> 2;             // staging row 0..63
    const int sc = (tid & 3) * 16;       // staging col 0,16,32,48

    v8f acc0 = {};
    v8f acc1 = {};

    for (int k0 = 0; k0 < K; k0 += 64) {
        const _Float16* ga = &A [(size_t)(m0 + sr) * K + k0 + sc];
        const _Float16* gw = &Wg[(size_t)(n0 + sr) * K + k0 + sc];
        *(h8*)&As[sr * 72 + sc]     = *(const h8*)ga;
        *(h8*)&As[sr * 72 + sc + 8] = *(const h8*)(ga + 8);
        *(h8*)&Ws[sr * 72 + sc]     = *(const h8*)gw;
        *(h8*)&Ws[sr * 72 + sc + 8] = *(const h8*)(gw + 8);
        __builtin_prefetch(ga + 64, 0, 0);   // global_prefetch_b8: next k-tile
        __builtin_prefetch(gw + 64, 0, 0);
        __syncthreads();

        const int arow = mi * 16 + l16;
        const int nr0  = (nia + 0) * 16 + l16;
        const int nr1  = (nia + 1) * 16 + l16;
#pragma unroll
        for (int kc = 0; kc < 64; kc += 32) {
            v16h a  = frag16(&As[arow * 72 + kc + 8 * half],
                             &As[arow * 72 + kc + 16 + 8 * half]);
            v16h b0 = frag16(&Ws[nr0 * 72 + kc + 8 * half],
                             &Ws[nr0 * 72 + kc + 16 + 8 * half]);
            v16h b1 = frag16(&Ws[nr1 * 72 + kc + 8 * half],
                             &Ws[nr1 * 72 + kc + 16 + 8 * half]);
            acc0 = __builtin_amdgcn_wmma_f32_16x16x32_f16(false, a, false, b0, (short)0, acc0, false, false);
            acc1 = __builtin_amdgcn_wmma_f32_16x16x32_f16(false, a, false, b1, (short)0, acc1, false, false);
        }
        __syncthreads();
    }

#pragma unroll
    for (int p = 0; p < 2; ++p) {
        v8f acc = p ? acc1 : acc0;
        int ni  = nia + p;
#pragma unroll
        for (int r = 0; r < 8; ++r) {
            int row  = m0 + mi * 16 + r + 8 * half;
            int colg = n0 + ni * 16 + l16;
            float v  = acc[r] + bias[colg];
            if constexpr (EPI == 0) {
                outh[(size_t)row * N + colg] = (_Float16)v;
            } else if constexpr (EPI == 1) {
                float gl = 0.5f * v * (1.0f + erff(v * 0.70710678118f));
                outh[(size_t)row * N + colg] = (_Float16)gl;
            } else if constexpr (EPI == 2) {
                int o = win_to_orig(row);
                outf[(size_t)o * CCH + colg] = resid[(size_t)o * CCH + colg] + v;
            } else {
                outf[(size_t)row * N + colg] = resid[(size_t)row * N + colg] + v;
            }
        }
    }
}

// ---------------------------------------------------------------------------
// Windowed attention, one block per (window, head). N=64 tokens, hd=32.
// ---------------------------------------------------------------------------
__global__ __launch_bounds__(128) void attn_kernel(
    const _Float16* __restrict__ qkv,       // [65536][576] (q|k|v, per-head 32)
    const float* __restrict__ bias_table,   // [343][6]
    _Float16* __restrict__ out)             // [65536][192]
{
    __shared__ __align__(16) _Float16 q_s[64 * 40];
    __shared__ __align__(16) _Float16 k_s[64 * 40];
    __shared__ __align__(16) _Float16 vT [32 * 72];   // vT[d][j]
    __shared__ __align__(16) float    S  [64 * 66];
    __shared__ __align__(16) _Float16 P  [64 * 72];
    __shared__ float bias_s[343 * 6];

    const int wi   = blockIdx.x / 6;     // global window 0..1023
    const int h    = blockIdx.x % 6;
    const int tid  = threadIdx.x;
    const int wave = tid >> 5;
    const int lane = tid & 31;
    const int half = lane >> 4;
    const int l16  = lane & 15;

    for (int i = tid; i < 343 * 6; i += 128) bias_s[i] = bias_table[i];

    {   // load q,k (row-major) and v (transposed): 16 halves each per thread
        int j  = tid >> 1;
        int d0 = (tid & 1) * 16;
        const _Float16* base = qkv + (size_t)(wi * 64 + j) * 576 + h * 32;
        *(h8*)&q_s[j * 40 + d0]     = *(const h8*)(base + d0);
        *(h8*)&q_s[j * 40 + d0 + 8] = *(const h8*)(base + d0 + 8);
        *(h8*)&k_s[j * 40 + d0]     = *(const h8*)(base + 192 + d0);
        *(h8*)&k_s[j * 40 + d0 + 8] = *(const h8*)(base + 192 + d0 + 8);
        const _Float16* vb = base + 384 + d0;
#pragma unroll
        for (int dd = 0; dd < 16; ++dd) vT[(d0 + dd) * 72 + j] = vb[dd];
    }
    __syncthreads();

    // S = (q . k^T) * scale  — wave handles row block mi=wave, all 4 ni
    {
        const float scale = 0.17677669529663687f;   // 1/sqrt(32)
        int arow = wave * 16 + l16;
        v16h a = frag16(&q_s[arow * 40 + 8 * half], &q_s[arow * 40 + 16 + 8 * half]);
#pragma unroll
        for (int ni = 0; ni < 4; ++ni) {
            int n = ni * 16 + l16;
            v16h b = frag16(&k_s[n * 40 + 8 * half], &k_s[n * 40 + 16 + 8 * half]);
            v8f c = {};
            c = __builtin_amdgcn_wmma_f32_16x16x32_f16(false, a, false, b, (short)0, c, false, false);
#pragma unroll
            for (int r = 0; r < 8; ++r)
                S[(wave * 16 + r + 8 * half) * 66 + ni * 16 + l16] = c[r] * scale;
        }
    }
    __syncthreads();

    // softmax rows (bias + shift-mask computed analytically), P = f16 probs
    if (tid < 64) {
        int i    = tid;
        int th_i = i >> 4, tw_i = (i >> 2) & 3, td_i = i & 3;
        int wb = wi & 511;
        int wh = wb >> 6, ww = (wb >> 3) & 7, wd = wb & 7;
        // region of shifted coordinate a = 4*widx + t : [0,28) [28,30) [30,32)
        auto reg = [](int widx, int t) { int a = widx * 4 + t; return a < 28 ? 0 : (a < 30 ? 1 : 2); };
        int rhi = reg(wh, th_i), rwi = reg(ww, tw_i), rdi = reg(wd, td_i);

        float mx = -1e30f;
        for (int j = 0; j < 64; ++j) {
            int th_j = j >> 4, tw_j = (j >> 2) & 3, td_j = j & 3;
            int rel  = (th_i - th_j + 3) * 49 + (tw_i - tw_j + 3) * 7 + (td_i - td_j + 3);
            bool same = (rhi == reg(wh, th_j)) && (rwi == reg(ww, tw_j)) && (rdi == reg(wd, td_j));
            float v = S[i * 66 + j] + bias_s[rel * 6 + h] + (same ? 0.f : -100.f);
            S[i * 66 + j] = v;
            mx = fmaxf(mx, v);
        }
        float sum = 0.f;
        for (int j = 0; j < 64; ++j) {
            float e = __expf(S[i * 66 + j] - mx);
            S[i * 66 + j] = e;
            sum += e;
        }
        float inv = 1.0f / sum;
        for (int j = 0; j < 64; ++j)
            P[i * 72 + j] = (_Float16)(S[i * 66 + j] * inv);
    }
    __syncthreads();

    // O = P . v  (M=64, N=32, K=64) — wave handles mi=wave, ni 0..1
    {
        v8f o0 = {};
        v8f o1 = {};
        int arow = wave * 16 + l16;
#pragma unroll
        for (int k0 = 0; k0 < 64; k0 += 32) {
            v16h a = frag16(&P[arow * 72 + k0 + 8 * half], &P[arow * 72 + k0 + 16 + 8 * half]);
            int n0i = 0 * 16 + l16;
            int n1i = 1 * 16 + l16;
            v16h b0 = frag16(&vT[n0i * 72 + k0 + 8 * half], &vT[n0i * 72 + k0 + 16 + 8 * half]);
            v16h b1 = frag16(&vT[n1i * 72 + k0 + 8 * half], &vT[n1i * 72 + k0 + 16 + 8 * half]);
            o0 = __builtin_amdgcn_wmma_f32_16x16x32_f16(false, a, false, b0, (short)0, o0, false, false);
            o1 = __builtin_amdgcn_wmma_f32_16x16x32_f16(false, a, false, b1, (short)0, o1, false, false);
        }
#pragma unroll
        for (int r = 0; r < 8; ++r) {
            int row = wave * 16 + r + 8 * half;
            size_t obase = (size_t)(wi * 64 + row) * CCH + h * 32;
            out[obase + 0  + l16] = (_Float16)o0[r];
            out[obase + 16 + l16] = (_Float16)o1[r];
        }
    }
}

// ---------------------------------------------------------------------------
extern "C" void kernel_launch(void* const* d_in, const int* in_sizes, int n_in,
                              void* d_out, int out_size, void* d_ws, size_t ws_size,
                              hipStream_t stream) {
    const float* x          = (const float*)d_in[0];
    const float* g1         = (const float*)d_in[1];
    const float* b1         = (const float*)d_in[2];
    const float* wqkv       = (const float*)d_in[3];
    const float* bqkv       = (const float*)d_in[4];
    const float* wo         = (const float*)d_in[5];
    const float* bo         = (const float*)d_in[6];
    const float* bias_table = (const float*)d_in[7];
    const float* g2         = (const float*)d_in[8];
    const float* b2         = (const float*)d_in[9];
    const float* w1         = (const float*)d_in[10];
    const float* bm1        = (const float*)d_in[11];
    const float* w2         = (const float*)d_in[12];
    const float* bm2        = (const float*)d_in[13];
    float* out              = (float*)d_out;

    // workspace partition
    char* p = (char*)d_ws;
    _Float16* xw    = (_Float16*)p;  p += (size_t)TOK * CCH * 2;   // 25 MB
    _Float16* qkv   = (_Float16*)p;  p += (size_t)TOK * 576 * 2;   // 75 MB
    _Float16* aout  = (_Float16*)p;  p += (size_t)TOK * CCH * 2;   // 25 MB
    float*    xo    = (float*)p;     p += (size_t)TOK * CCH * 4;   // 50 MB
    _Float16* xn2   = (_Float16*)p;  p += (size_t)TOK * CCH * 2;   // 25 MB
    _Float16* h1    = (_Float16*)p;  p += (size_t)TOK * 768 * 2;   // 100 MB
    _Float16* wqkvT = (_Float16*)p;  p += (size_t)192 * 576 * 2;   // f16 [N][K]
    _Float16* woT   = (_Float16*)p;  p += (size_t)192 * 192 * 2;
    _Float16* w1T   = (_Float16*)p;  p += (size_t)192 * 768 * 2;
    _Float16* w2T   = (_Float16*)p;

    // 0) one-shot weight conversion to f16 transposed [N][K]
    wconv_kernel<<<(192 * 576 + 255) / 256, 256, 0, stream>>>(wqkv, wqkvT, 192, 576);
    wconv_kernel<<<(192 * 192 + 255) / 256, 256, 0, stream>>>(wo,   woT,   192, 192);
    wconv_kernel<<<(192 * 768 + 255) / 256, 256, 0, stream>>>(w1,   w1T,   192, 768);
    wconv_kernel<<<(768 * 192 + 255) / 256, 256, 0, stream>>>(w2,   w2T,   768, 192);

    // 1) LN1 + shift + window-partition -> f16 (windowed token order)
    ln_kernel<<<TOK, 32, 0, stream>>>(x, g1, b1, xw, 1);

    // 2) QKV GEMM: [65536,192] x [192,576]
    gemm64x64<0><<<dim3(TOK / 64, 576 / 64), 256, 0, stream>>>(
        xw, wqkvT, bqkv, qkv, nullptr, nullptr, 192, 576);

    // 3) windowed attention (1024 windows x 6 heads)
    attn_kernel<<<1024 * 6, 128, 0, stream>>>(qkv, bias_table, aout);

    // 4) out projection + window-reverse + residual -> xo (f32, natural order)
    gemm64x64<2><<<dim3(TOK / 64, 192 / 64), 256, 0, stream>>>(
        aout, woT, bo, nullptr, xo, x, 192, 192);

    // 5) LN2 -> f16
    ln_kernel<<<TOK, 32, 0, stream>>>(xo, g2, b2, xn2, 0);

    // 6) MLP fc1 + GELU: [65536,192] x [192,768]
    gemm64x64<1><<<dim3(TOK / 64, 768 / 64), 256, 0, stream>>>(
        xn2, w1T, bm1, h1, nullptr, nullptr, 192, 768);

    // 7) MLP fc2 + residual(xo) -> d_out: [65536,768] x [768,192]
    gemm64x64<3><<<dim3(TOK / 64, 192 / 64), 256, 0, stream>>>(
        h1, w2T, bm2, nullptr, out, xo, 768, 192);
}
